// CrossAttentionPairBias_30588757082754
// MI455X (gfx1250) — matrix operand order repaired and checked by reading, not compiled, not run on hardware
//
#include <hip/hip_runtime.h>
#include <hip/hip_bf16.h>
#include <math.h>

typedef __attribute__((ext_vector_type(16))) _Float16 v16h;
typedef __attribute__((ext_vector_type(8)))  _Float16 v8h;
typedef __attribute__((ext_vector_type(8)))  float    v8f;
typedef __attribute__((ext_vector_type(4)))  unsigned int u32x4;
typedef __attribute__((ext_vector_type(8)))  unsigned int u32x8;

#define B_   4
#define N_   8192
#define C_   128
#define CZ_  16
#define H_   4
#define CH_  32
#define NQ_  32
#define NK_  128
#define NB_  (N_/NQ_)
#define INFV 1000000000.0f
#define EPS_ 1e-5f
#define WSWZ 16384   // halfs per repacked weight: 8 ntiles * 4 kchunks * 32 lanes * 16

__device__ __forceinline__ float sig_(float x){ return 1.0f/(1.0f+expf(-x)); }

__device__ __forceinline__ v8f wmma_(v16h a, v16h b, v8f c){
  return __builtin_amdgcn_wmma_f32_16x16x32_f16(false, a, false, b, (short)0, c, false, false);
}

// ---- fragment loaders (this code's k order: contiguous, k = 16g+e in both A and B) -----------
// A 16x32 f16 from row-major LDS at (m0,k0): lane m=l&15,g=l>>4;
// per-lane data = [16g..16g+15] -> two 16B loads.
__device__ __forceinline__ v16h ldA_h(const _Float16* p, int ld){
  int l = threadIdx.x & 31, m = l & 15, g = l >> 4;
  const _Float16* q = p + m * ld + (g << 4);
  v8h lo = *(const v8h*)(q);
  v8h hi = *(const v8h*)(q + 8);
  return __builtin_shufflevector(lo, hi, 0,1,2,3,4,5,6,7,8,9,10,11,12,13,14,15);
}
// A 16x32 from f32 LDS (attn weights), converted elementwise
__device__ __forceinline__ v16h ldA_f(const float* p, int ld){
  int l = threadIdx.x & 31, m = l & 15, g = l >> 4;
  v16h a;
#pragma unroll
  for (int e = 0; e < 16; ++e){
    int r = e >> 1, pb = e & 1;
    int k = (g << 4) + (r << 1) + pb;
    a[e] = (_Float16)p[m * ld + k];
  }
  return a;
}
// B 32x16, B(k,n) sourced from N-major storage: p at (n0,k0) of [N][K] array;
// lane n=l&15, elems = K = 16g+e -> contiguous 16 halfs per lane.
__device__ __forceinline__ v16h ldB_t(const _Float16* p, int ld){
  int l = threadIdx.x & 31, n = l & 15, g = l >> 4;
  const _Float16* q = p + n * ld + (g << 4);
  v8h lo = *(const v8h*)(q);
  v8h hi = *(const v8h*)(q + 8);
  return __builtin_shufflevector(lo, hi, 0,1,2,3,4,5,6,7,8,9,10,11,12,13,14,15);
}
// B fragment from pre-swizzled f16 weights: one 32B load per lane.
__device__ __forceinline__ v16h ldB_sw(const _Float16* swz, int nt, int kk){
  int l = threadIdx.x & 31;
  return *(const v16h*)(swz + ((((nt << 2) + kk) << 5) + l) * 16);
}

// ---- TDM: 1-row tile, 8B elements, pad 16B after every 256B (stride 272B) --
__device__ __forceinline__ void tdm_load_padded(unsigned int lds_off, const void* gptr,
                                                unsigned int n8){
  unsigned long long ga = (unsigned long long)(size_t)gptr;
  u32x4 g0;
  g0[0] = 1u;                                     // count=1, user descriptor
  g0[1] = lds_off;                                // lds_addr
  g0[2] = (unsigned int)ga;                       // global_addr[31:0]
  g0[3] = (unsigned int)((ga >> 32) & 0x1FFFFFFu) // global_addr[56:32]
        | (2u << 30);                             // type=2 (image)
  u32x8 g1;
  g1[0] = (3u << 16)    // data_size = 8B
        | (1u << 20)    // pad_enable
        | (5u << 22)    // pad_interval: 64 DWORDs (256B)
        | (3u << 25);   // pad_amount: 4 DWORDs (16B)
  g1[1] = (n8 & 0xFFFFu) << 16;          // tensor_dim0[15:0] @ bits79:48
  g1[2] = ((n8 >> 16) & 0xFFFFu)         // tensor_dim0[31:16]
        | (1u << 16);                    // tensor_dim1 = 1
  g1[3] = (n8 & 0xFFFFu) << 16;          // tile_dim0 @ bits127:112
  g1[4] = 1u;                            // tile_dim1 = 1, tile_dim2 = 0
  g1[5] = n8;                            // tensor_dim0_stride[31:0]
  g1[6] = 0u;
  g1[7] = 0u;
  asm volatile("tensor_load_to_lds %0, %1" :: "s"(g0), "s"(g1) : "memory");
}

// ---------------- Kernel 0: weight repack to f16 fragment order -------------
struct RepackArgs { const float* src[10]; _Float16* dst; };

__global__ void __launch_bounds__(256) repack_kernel(RepackArgs R){
  const float* W = R.src[blockIdx.y];
  _Float16* out = R.dst + (size_t)blockIdx.y * WSWZ;
  int o = blockIdx.x * 256 + threadIdx.x;   // grid.x = WSWZ/256 = 64
  int e  = o & 15;
  int l  = (o >> 4) & 31;
  int kk = (o >> 9) & 3;
  int nt = o >> 11;
  int n = (nt << 4) + (l & 15);
  int k = (kk << 5) + ((l >> 4) << 4) + e;
  out[o] = (_Float16)W[k * C_ + n];
}

// ---------------- Kernel 1: per-position adaptive-LN + projections ----------
struct ProjArgs {
  const float *a, *s;
  const float *ln_s_q_w, *bg_q, *ln_s_k_w, *bg_k, *bg, *b_ado;
  const _Float16 *wWg_q, *wWb_q, *wWg_k, *wWb_k, *wWq, *wWg, *wWk, *wWv, *wW_ado;
  _Float16 *qS, *kS, *vS, *gS, *adoS;
};

__device__ __forceinline__ void gemm_lds(const _Float16 (*Ah)[136], const _Float16* swz,
                                         float (*Out)[132], int wave){
  for (int tt = wave; tt < 16; tt += 8){
    int m0 = (tt >> 3) << 4, nt = tt & 7;
    v8f acc = {};
#pragma unroll
    for (int kk = 0; kk < 4; ++kk)
      acc = wmma_(ldA_h(&Ah[m0][kk*32], 136), ldB_sw(swz, nt, kk), acc);
    int l = threadIdx.x & 31, n = l & 15, g = l >> 4;
#pragma unroll
    for (int r = 0; r < 8; ++r) Out[m0 + r + 8*g][(nt << 4) + n] = acc[r];
  }
}
// mode 0: x*scale ; mode 1: sigmoid(x + bias[n])
__device__ __forceinline__ void gemm_glb(const _Float16 (*Ah)[136], const _Float16* swz,
                                         _Float16* dst, size_t row0, const float* bias,
                                         float scale, int mode, int wave){
  for (int tt = wave; tt < 16; tt += 8){
    int m0 = (tt >> 3) << 4, nt = tt & 7;
    v8f acc = {};
#pragma unroll
    for (int kk = 0; kk < 4; ++kk)
      acc = wmma_(ldA_h(&Ah[m0][kk*32], 136), ldB_sw(swz, nt, kk), acc);
    int l = threadIdx.x & 31, n = l & 15, g = l >> 4;
    int n0 = nt << 4;
#pragma unroll
    for (int r = 0; r < 8; ++r){
      float x = acc[r];
      if (mode == 0) x *= scale;
      else           x = sig_(x + bias[n0 + n]);
      dst[(row0 + (size_t)(m0 + r + 8*g)) * C_ + n0 + n] = (_Float16)x;
    }
  }
}

__global__ void __launch_bounds__(256) proj_kernel(ProjArgs A){
  __shared__ __attribute__((aligned(32))) float    sNA[32][132];
  __shared__ __attribute__((aligned(32))) float    sF0[32][132];
  __shared__ __attribute__((aligned(32))) float    sF1[32][132];
  __shared__ __attribute__((aligned(32))) _Float16 sQN[32][136];
  __shared__ __attribute__((aligned(32))) _Float16 sKN[32][136];
  __shared__ __attribute__((aligned(32))) _Float16 sSR[32][136];

  const int t = threadIdx.x, wave = t >> 5;
  const size_t row0 = (size_t)blockIdx.x * 32;

  for (int i = t; i < 32*32; i += 256){
    int r = i >> 5, c4 = (i & 31) << 2;
    float4 av = *(const float4*)(A.a + (row0 + r)*C_ + c4);
    float4 sv = *(const float4*)(A.s + (row0 + r)*C_ + c4);
    sNA[r][c4+0]=av.x; sNA[r][c4+1]=av.y; sNA[r][c4+2]=av.z; sNA[r][c4+3]=av.w;
    sF0[r][c4+0]=sv.x; sF0[r][c4+1]=sv.y; sF0[r][c4+2]=sv.z; sF0[r][c4+3]=sv.w;
  }
  __syncthreads();

  if (t < 32){
    float ma=0, qa=0, ms=0, qs=0;
    for (int c = 0; c < C_; ++c){
      float x = sNA[t][c]; ma += x; qa += x*x;
      float y = sF0[t][c]; ms += y; qs += y*y;
    }
    ma *= (1.0f/C_); ms *= (1.0f/C_);
    float ra = rsqrtf(fmaxf(qa*(1.0f/C_) - ma*ma, 0.f) + EPS_);
    float rs = rsqrtf(fmaxf(qs*(1.0f/C_) - ms*ms, 0.f) + EPS_);
    for (int c = 0; c < C_; ++c){
      float sraw = sF0[t][c];
      float na = (sNA[t][c]-ma)*ra;
      float ns = (sraw-ms)*rs;
      sNA[t][c] = na;
      sQN[t][c] = (_Float16)(ns * A.ln_s_q_w[c]);
      sKN[t][c] = (_Float16)(ns * A.ln_s_k_w[c]);
      sSR[t][c] = (_Float16)sraw;
    }
  }
  __syncthreads();

  gemm_lds(sQN, A.wWg_q, sF0, wave);
  gemm_lds(sQN, A.wWb_q, sF1, wave);
  __syncthreads();
  for (int i = t; i < 32*C_; i += 256){
    int r = i >> 7, c = i & 127;
    sQN[r][c] = (_Float16)(sig_(sF0[r][c] + A.bg_q[c]) * sNA[r][c] + sF1[r][c]); // aq
  }
  __syncthreads();

  gemm_lds(sKN, A.wWg_k, sF0, wave);
  gemm_lds(sKN, A.wWb_k, sF1, wave);
  __syncthreads();
  for (int i = t; i < 32*C_; i += 256){
    int r = i >> 7, c = i & 127;
    sKN[r][c] = (_Float16)(sig_(sF0[r][c] + A.bg_k[c]) * sNA[r][c] + sF1[r][c]); // ak
  }
  __syncthreads();

  const float scale = 0.17677669529663687f;  // 1/sqrt(CH)
  gemm_glb(sQN, A.wWq,    A.qS,   row0, nullptr,  scale, 0, wave);
  gemm_glb(sQN, A.wWg,    A.gS,   row0, A.bg,     1.f,   1, wave);
  gemm_glb(sKN, A.wWk,    A.kS,   row0, nullptr,  1.f,   0, wave);
  gemm_glb(sKN, A.wWv,    A.vS,   row0, nullptr,  1.f,   0, wave);
  gemm_glb(sSR, A.wW_ado, A.adoS, row0, A.b_ado,  1.f,   1, wave);
}

// ---------------- Kernel 2: windowed attention with pair bias ---------------
struct AttnArgs {
  const float *z, *mask, *ln_z_w, *ln_z_b, *W_z, *bo;
  const _Float16 *qS, *kS, *vS, *gS, *adoS, *wWo;
  float* out;
};

__global__ void __launch_bounds__(256) attn_kernel(AttnArgs A){
  __shared__ __attribute__((aligned(32))) _Float16 sK [128][136]; // [key][ch]
  __shared__ __attribute__((aligned(32))) _Float16 sVT[128][136]; // [ch][key]
  __shared__ __attribute__((aligned(32))) _Float16 sQ [32][136];  // q -> gated o
  __shared__ float sL[H_][32][132];                               // logits/attn
  __shared__ float sZW[CZ_], sZB[CZ_], sWZ[CZ_][H_];

  const int t = threadIdx.x, wave = t >> 5;
  const int blk = blockIdx.x, b = blockIdx.y;
  const size_t rowq0 = (size_t)b * N_ + (size_t)blk * NQ_;
  const bool interior = (blk >= 2) && (blk <= NB_ - 3);  // full window in range

  if (t < CZ_){ sZW[t] = A.ln_z_w[t]; sZB[t] = A.ln_z_b[t]; }
  if (t < CZ_*H_) sWZ[t >> 2][t & 3] = A.W_z[t];

  // --- async tensor staging (wave 0 issues; DMA overlaps zb stage) ---
  if (t < 32){
    tdm_load_padded((unsigned int)(size_t)&sQ[0][0], A.qS + rowq0 * C_, 32*C_/4);
    if (interior){
      size_t rowk0 = (size_t)b * N_ + (size_t)blk * NQ_ - (NK_-NQ_)/2;
      tdm_load_padded((unsigned int)(size_t)&sK[0][0], A.kS + rowk0 * C_, NK_*C_/4);
    }
  }
  if (!interior){
    for (int i = t; i < NK_*C_; i += 256){
      int j = i >> 7, c = i & 127;
      int pos = blk*NQ_ - (NK_-NQ_)/2 + j;
      int pc = pos < 0 ? 0 : (pos > N_-1 ? N_-1 : pos);
      sK[j][c] = A.kS[((size_t)b*N_ + pc)*C_ + c];
    }
  }
  // v staged transposed [ch][key] (generic path; scattered LDS writes)
  for (int i = t; i < NK_*C_; i += 256){
    int j = i >> 7, c = i & 127;
    int pos = blk*NQ_ - (NK_-NQ_)/2 + j;
    int pc = pos < 0 ? 0 : (pos > N_-1 ? N_-1 : pos);
    sVT[c][j] = A.vS[((size_t)b*N_ + pc)*C_ + c];
  }

  // --- pair bias: LN(z)@W_z + INF*(mask-1) -> sL (overlaps TDM) ---
  const size_t zbase = (((size_t)b*NB_ + blk)*NQ_) * (size_t)NK_ * CZ_;
  for (int i = t; i < NQ_*NK_; i += 256){
    int q = i >> 7, j = i & 127;
    const float* zp = A.z + zbase + ((size_t)q*NK_ + j)*CZ_;
    __builtin_prefetch(zp + 256*CZ_, 0, 0);   // next line this thread streams
    float m = 0;
#pragma unroll
    for (int c = 0; c < CZ_; ++c) m += zp[c];
    m *= (1.0f/CZ_);
    float var = 0;
#pragma unroll
    for (int c = 0; c < CZ_; ++c){ float d = zp[c]-m; var += d*d; }
    float rs = rsqrtf(var*(1.0f/CZ_) + EPS_);
    float acc[H_] = {0,0,0,0};
#pragma unroll
    for (int c = 0; c < CZ_; ++c){
      float y = (zp[c]-m)*rs*sZW[c] + sZB[c];
#pragma unroll
      for (int h = 0; h < H_; ++h) acc[h] += y * sWZ[c][h];
    }
    int pos = blk*NQ_ - (NK_-NQ_)/2 + j;
    float km = (pos >= 0 && pos < N_) ? A.mask[(size_t)b*N_ + pos] : 0.f;
    float qm = A.mask[rowq0 + q];
    float bias = INFV * (qm*km - 1.0f);
#pragma unroll
    for (int h = 0; h < H_; ++h) sL[h][q][j] = acc[h] + bias;
  }
  if (t < 32) __builtin_amdgcn_s_wait_tensorcnt(0);
  __syncthreads();

  // --- logits += q @ k^T : wave <-> (head, m-tile) ---
  {
    int h = wave >> 1, m0 = (wave & 1) << 4;
    int l = t & 31, n = l & 15, g = l >> 4;
    for (int nt = 0; nt < 8; ++nt){
      int n0 = nt << 4;
      v8f acc;
#pragma unroll
      for (int r = 0; r < 8; ++r) acc[r] = sL[h][m0 + r + 8*g][n0 + n];
      acc = wmma_(ldA_h(&sQ[m0][h*CH_], 136), ldB_t(&sK[n0][h*CH_], 136), acc);
#pragma unroll
      for (int r = 0; r < 8; ++r) sL[h][m0 + r + 8*g][n0 + n] = acc[r];
    }
  }
  __syncthreads();

  // --- softmax over keys (thread per (h,q) row) ---
  if (t < H_*NQ_){
    int h = t >> 5, q = t & 31;
    float mx = -3.4e38f;
    for (int j = 0; j < NK_; ++j) mx = fmaxf(mx, sL[h][q][j]);
    float sum = 0.f;
    for (int j = 0; j < NK_; ++j){ float e = expf(sL[h][q][j] - mx); sL[h][q][j] = e; sum += e; }
    float inv = 1.0f/sum;
    for (int j = 0; j < NK_; ++j) sL[h][q][j] *= inv;
  }
  __syncthreads();

  // --- o = attn @ v, gated by g -> sQ (reuse) ---
  {
    int h = wave >> 1, m0 = (wave & 1) << 4;
    int l = t & 31, n = l & 15, g = l >> 4;
    for (int nt = 0; nt < 2; ++nt){
      int n0 = nt << 4;
      v8f acc = {};
#pragma unroll
      for (int kk = 0; kk < 4; ++kk)
        acc = wmma_(ldA_f(&sL[h][m0][kk*32], 132), ldB_t(&sVT[h*CH_ + n0][kk*32], 136), acc);
#pragma unroll
      for (int r = 0; r < 8; ++r){
        int m = m0 + r + 8*g;
        int c = h*CH_ + n0 + n;
        float gt = (float)A.gS[(rowq0 + m)*C_ + c];
        sQ[m][c] = (_Float16)(acc[r] * gt);
      }
    }
  }
  __syncthreads();

  // --- out = (o @ Wo + bo) * ado ---
  for (int tt = wave; tt < 16; tt += 8){
    int m0 = (tt >> 3) << 4, nt = tt & 7, n0 = nt << 4;
    v8f acc = {};
#pragma unroll
    for (int kk = 0; kk < 4; ++kk)
      acc = wmma_(ldA_h(&sQ[m0][kk*32], 136), ldB_sw(A.wWo, nt, kk), acc);
    int l = t & 31, n = l & 15, g = l >> 4;
#pragma unroll
    for (int r = 0; r < 8; ++r){
      int m = m0 + r + 8*g;
      float x = acc[r] + A.bo[n0 + n];
      float ad = (float)A.adoS[(rowq0 + m)*C_ + n0 + n];
      A.out[(rowq0 + m)*C_ + n0 + n] = ad * x;
    }
  }
}

// ---------------- host launch ----------------
extern "C" void kernel_launch(void* const* d_in, const int* in_sizes, int n_in,
                              void* d_out, int out_size, void* d_ws, size_t ws_size,
                              hipStream_t stream){
  (void)in_sizes; (void)n_in; (void)out_size; (void)ws_size;
  const float* a        = (const float*)d_in[0];
  const float* z        = (const float*)d_in[1];
  const float* s        = (const float*)d_in[2];
  const float* mask     = (const float*)d_in[3];
  const float* ln_s_q_w = (const float*)d_in[4];
  const float* Wg_q     = (const float*)d_in[5];
  const float* bg_q     = (const float*)d_in[6];
  const float* Wb_q     = (const float*)d_in[7];
  const float* ln_s_k_w = (const float*)d_in[8];
  const float* Wg_k     = (const float*)d_in[9];
  const float* bg_k     = (const float*)d_in[10];
  const float* Wb_k     = (const float*)d_in[11];
  const float* ln_z_w   = (const float*)d_in[12];
  const float* ln_z_b   = (const float*)d_in[13];
  const float* W_z      = (const float*)d_in[14];
  const float* Wq       = (const float*)d_in[15];
  const float* Wk       = (const float*)d_in[16];
  const float* Wv       = (const float*)d_in[17];
  const float* Wg       = (const float*)d_in[18];
  const float* bg       = (const float*)d_in[19];
  const float* Wo       = (const float*)d_in[20];
  const float* bo       = (const float*)d_in[21];
  const float* W_ado    = (const float*)d_in[22];
  const float* b_ado    = (const float*)d_in[23];

  size_t P = (size_t)B_ * N_ * C_;
  _Float16* qS   = (_Float16*)d_ws;
  _Float16* kS   = qS + P;
  _Float16* vS   = kS + P;
  _Float16* gS   = vS + P;
  _Float16* adoS = gS + P;
  _Float16* wS   = adoS + P;   // 10 * WSWZ halfs of repacked weights

  RepackArgs ra;
  ra.src[0]=Wg_q; ra.src[1]=Wb_q; ra.src[2]=Wg_k; ra.src[3]=Wb_k;
  ra.src[4]=Wq;   ra.src[5]=Wg;   ra.src[6]=Wk;   ra.src[7]=Wv;
  ra.src[8]=W_ado; ra.src[9]=Wo;  ra.dst = wS;
  repack_kernel<<<dim3(WSWZ/256, 10), 256, 0, stream>>>(ra);

  ProjArgs pa;
  pa.a=a; pa.s=s; pa.ln_s_q_w=ln_s_q_w; pa.bg_q=bg_q; pa.ln_s_k_w=ln_s_k_w;
  pa.bg_k=bg_k; pa.bg=bg; pa.b_ado=b_ado;
  pa.wWg_q=wS+0*WSWZ; pa.wWb_q=wS+1*WSWZ; pa.wWg_k=wS+2*WSWZ; pa.wWb_k=wS+3*WSWZ;
  pa.wWq=wS+4*WSWZ;   pa.wWg=wS+5*WSWZ;   pa.wWk=wS+6*WSWZ;   pa.wWv=wS+7*WSWZ;
  pa.wW_ado=wS+8*WSWZ;
  pa.qS=qS; pa.kS=kS; pa.vS=vS; pa.gS=gS; pa.adoS=adoS;
  proj_kernel<<<(B_*N_)/32, 256, 0, stream>>>(pa);

  AttnArgs aa;
  aa.z=z; aa.mask=mask; aa.ln_z_w=ln_z_w; aa.ln_z_b=ln_z_b; aa.W_z=W_z; aa.bo=bo;
  aa.qS=qS; aa.kS=kS; aa.vS=vS; aa.gS=gS; aa.adoS=adoS; aa.wWo=wS+9*WSWZ;
  aa.out=(float*)d_out;
  attn_kernel<<<dim3(NB_, B_), 256, 0, stream>>>(aa);
}
